// AttentionModel_73186242723846
// MI455X (gfx1250) — matrix operand order repaired and checked by reading, not compiled, not run on hardware
//
#include <hip/hip_runtime.h>
#include <stdint.h>

typedef uint16_t u16;
typedef __attribute__((ext_vector_type(16))) __bf16 v16bf;
typedef __attribute__((ext_vector_type(8)))  float  v8f;
typedef __attribute__((ext_vector_type(4)))  unsigned int u32x4;

// ---------------- constants ----------------
#define L_TOK   8192      // tokens per sentence
#define EMB     300
#define S_EMB   320       // padded emb stride (mult of 32)
#define S_F     256       // padded 200-dim stride
#define S_CAT   640       // padded 600-dim stride
#define NPAD    256       // padded N rows for weight Bt buffers
#define FK_LDS  264       // LDS row stride (u16) for K-major staging (bank-conflict-free)
#define VT_LDS  40        // LDS row stride (u16) for Vt staging

struct FragU { union { v16bf v; u32x4 q[2]; }; };

__device__ __forceinline__ u16 f2bf(float x) {
    unsigned u = __float_as_uint(x);
    return (u16)((u + 0x7FFFu + ((u >> 16) & 1u)) >> 16);
}

// A fragment: 16x32 bf16 from row-major [rows x ld] at (row0, kk)
__device__ __forceinline__ FragU load_frag_a(const u16* p, int row0, int kk, int ld, int lane) {
    int half = lane >> 4, l15 = lane & 15;
    int r  = row0 + l15;
    int k0 = kk + (half ? 16 : 0);
    int k1 = kk + (half ? 24 : 8);
    FragU f;
    f.q[0] = *(const u32x4*)(p + (size_t)r * ld + k0);
    f.q[1] = *(const u32x4*)(p + (size_t)r * ld + k1);
    return f;
}

// B fragment: 32x16 bf16; operand given as Bt row-major [N x K] at (n0, kk)
__device__ __forceinline__ FragU load_frag_b(const u16* p, int n0, int kk, int ld, int lane) {
    int half = lane >> 4, l15 = lane & 15;
    int n = n0 + l15;
    int k = kk + half * 16;
    const u32x4* q = (const u32x4*)(p + (size_t)n * ld + k);
    FragU f; f.q[0] = q[0]; f.q[1] = q[1];
    return f;
}

// B fragment from an LDS-staged tile [rows x ldRow(u16)], tile col base kcol
__device__ __forceinline__ FragU lds_frag_b(const u16* base, int n0, int kcol, int ldRow, int lane) {
    int half = lane >> 4, l15 = lane & 15;
    const u32x4* p = (const u32x4*)(base + (size_t)(n0 + l15) * ldRow + kcol + half * 16);
    FragU f; f.q[0] = p[0]; f.q[1] = p[1];
    return f;
}

__device__ __forceinline__ v8f wmma_bf16(const FragU& a, const FragU& b, v8f c) {
    return __builtin_amdgcn_wmma_f32_16x16x32_bf16(false, a.v, false, b.v, (short)0, c, false, false);
}

// ---------------- utility kernels ----------------
__global__ void zero_f32(float* p, int n) {
    int i = blockIdx.x * blockDim.x + threadIdx.x;
    if (i < n) p[i] = 0.f;
}

// f32 [R x C] -> bf16 [Rp x Cp], zero padded
__global__ void cast_pad(const float* __restrict__ in, u16* __restrict__ out,
                         int R, int C, int Cp, int total) {
    int idx = blockIdx.x * blockDim.x + threadIdx.x;
    if (idx >= total) return;
    int r = idx / Cp, c = idx - r * Cp;
    out[idx] = (r < R && c < C) ? f2bf(in[(size_t)r * C + c]) : (u16)0;
}

// f32 [Rin x Cin] -> bf16 transpose [Cp x Rin], zero padded rows
__global__ void transpose_cast(const float* __restrict__ in, u16* __restrict__ out,
                               int Rin, int Cin, int total) {
    int idx = blockIdx.x * blockDim.x + threadIdx.x;
    if (idx >= total) return;
    int n = idx / Rin, i = idx - n * Rin;
    out[idx] = (n < Cin) ? f2bf(in[(size_t)i * Cin + n]) : (u16)0;
}

// cat[i][0:300]=sen_bf16, [300:600]=attn(f32->bf16), [600:640]=0
__global__ void concat_kernel(const u16* __restrict__ senb, const float* __restrict__ att,
                              u16* __restrict__ cat) {
    int i = blockIdx.x, t = threadIdx.x;
    u16 v;
    if (t < EMB)           v = senb[(size_t)i * S_EMB + t];
    else if (t < 2 * EMB)  v = f2bf(att[(size_t)i * S_EMB + (t - EMB)]);
    else                   v = 0;
    cat[(size_t)i * S_CAT + t] = v;
}

// ---------------- generic bf16 GEMM + bias + relu ----------------
// C[M x ldc] = relu(A[M x Kp] * Bt[NPAD x Kp]^T + bias) ; Bt zero-padded beyond Ntrue
template<bool OUT_BF16>
__global__ __launch_bounds__(256) void gemm_bias_relu(
        const u16* __restrict__ A, const u16* __restrict__ Bt,
        const float* __restrict__ bias, void* __restrict__ out,
        int Kp, int lda, int ldb, int ldc, int Ntrue) {
    int wave = threadIdx.x >> 5, lane = threadIdx.x & 31;
    int half = lane >> 4, l15 = lane & 15;
    int row0 = blockIdx.x * 128 + wave * 16;
    int cb   = blockIdx.y * 32;

    v8f acc0, acc1;
    #pragma unroll
    for (int e = 0; e < 8; ++e) { acc0[e] = 0.f; acc1[e] = 0.f; }

    for (int kk = 0; kk < Kp; kk += 32) {
        FragU a  = load_frag_a(A, row0, kk, lda, lane);
        FragU b0 = load_frag_b(Bt, cb,      kk, ldb, lane);
        FragU b1 = load_frag_b(Bt, cb + 16, kk, ldb, lane);
        acc0 = wmma_bf16(a, b0, acc0);
        acc1 = wmma_bf16(a, b1, acc1);
    }

    int col0 = cb + l15, col1 = cb + 16 + l15;
    float bv0 = (col0 < Ntrue) ? bias[col0] : 0.f;
    float bv1 = (col1 < Ntrue) ? bias[col1] : 0.f;
    u16*   o16 = (u16*)out;
    float* o32 = (float*)out;
    #pragma unroll
    for (int r = 0; r < 8; ++r) {
        int row = row0 + r + half * 8;
        float v0 = fmaxf(acc0[r] + bv0, 0.f);
        float v1 = fmaxf(acc1[r] + bv1, 0.f);
        if (OUT_BF16) {
            o16[(size_t)row * ldc + col0] = f2bf(v0);
            o16[(size_t)row * ldc + col1] = f2bf(v1);
        } else {
            o32[(size_t)row * ldc + col0] = v0;
            o32[(size_t)row * ldc + col1] = v1;
        }
    }
}

// ---------------- softmax stats (online, flash pass 1) ----------------
// Row stats of E = Fq * Fk^T. Fk tiles are LDS-staged once per block (8x reuse).
__global__ __launch_bounds__(256) void softmax_stats(
        const u16* __restrict__ Fq, const u16* __restrict__ Fk,
        float* __restrict__ mOut, float* __restrict__ lOut) {
    int wave = threadIdx.x >> 5, lane = threadIdx.x & 31;
    int half = lane >> 4, l15 = lane & 15;
    int row0 = blockIdx.x * 128 + wave * 16;

    __shared__ alignas(16) u16 sFk[32 * FK_LDS];   // 32 rows of Fk, K=256, padded stride

    FragU af[8];
    #pragma unroll
    for (int k8 = 0; k8 < 8; ++k8) af[k8] = load_frag_a(Fq, row0, k8 * 32, S_F, lane);

    float m[8], l[8];
    #pragma unroll
    for (int r = 0; r < 8; ++r) { m[r] = -3.0e38f; l[r] = 0.f; }

    for (int j = 0; j < L_TOK; j += 32) {
        __syncthreads();   // previous-iteration readers done
        // stage Fk rows j..j+31, cols 0..255 (b128 per thread x4)
        #pragma unroll
        for (int q = 0; q < 4; ++q) {
            int e = threadIdx.x + q * 256;        // chunk id, 8 u16 each
            int row = e >> 5, chunk = e & 31;
            *(u32x4*)&sFk[row * FK_LDS + chunk * 8] =
                *(const u32x4*)(Fk + (size_t)(j + row) * S_F + chunk * 8);
        }
        __syncthreads();

        v8f s0, s1;
        #pragma unroll
        for (int e = 0; e < 8; ++e) { s0[e] = 0.f; s1[e] = 0.f; }
        #pragma unroll
        for (int k8 = 0; k8 < 8; ++k8) {
            FragU b0 = lds_frag_b(sFk, 0,  k8 * 32, FK_LDS, lane);
            FragU b1 = lds_frag_b(sFk, 16, k8 * 32, FK_LDS, lane);
            s0 = wmma_bf16(af[k8], b0, s0);
            s1 = wmma_bf16(af[k8], b1, s1);
        }
        #pragma unroll
        for (int r = 0; r < 8; ++r) {
            float v0 = s0[r], v1 = s1[r];
            float mn = fmaxf(m[r], fmaxf(v0, v1));
            l[r] = l[r] * __expf(m[r] - mn) + __expf(v0 - mn) + __expf(v1 - mn);
            m[r] = mn;
        }
    }
    // merge across the 16 lanes holding each row
    #pragma unroll
    for (int mask = 1; mask < 16; mask <<= 1) {
        #pragma unroll
        for (int r = 0; r < 8; ++r) {
            float mo = __shfl_xor(m[r], mask, 32);
            float lo = __shfl_xor(l[r], mask, 32);
            float mn = fmaxf(m[r], mo);
            l[r] = l[r] * __expf(m[r] - mn) + lo * __expf(mo - mn);
            m[r] = mn;
        }
    }
    if (l15 == 0) {
        int base = row0 + half * 8;
        #pragma unroll
        for (int r = 0; r < 8; ++r) { mOut[base + r] = m[r]; lOut[base + r] = l[r]; }
    }
}

// ---------------- flash pass 2: out = softmax(Fq*Fk^T) * V ----------------
// Vt [S_EMB x L_TOK] bf16 (rows >= EMB zero). out: [L_TOK x S_EMB] f32.
// Block: 64 rows (4 row-tiles x 2 col-halves). Fk & Vt panels LDS-staged per block.
__global__ __launch_bounds__(256) void attend_kernel(
        const u16* __restrict__ Fq, const u16* __restrict__ Fk,
        const float* __restrict__ mrow, const float* __restrict__ lrow,
        const u16* __restrict__ Vt, float* __restrict__ out) {
    int wave = threadIdx.x >> 5, lane = threadIdx.x & 31;
    int half = lane >> 4, l15 = lane & 15;
    int rowTile = wave >> 1, nHalf = wave & 1;
    int row0 = blockIdx.x * 64 + rowTile * 16;
    int nbase = nHalf * 160;

    __shared__ alignas(16) u16 sFk[32 * FK_LDS];     // ~16.9 KB
    __shared__ alignas(16) u16 sVt[320 * VT_LDS];    // ~25.6 KB
    __shared__ alignas(16) u16 sP[8][512];           // 8 KB (per-wave P tiles)

    // Fq A-fragments stay resident in VGPRs for the whole kernel
    FragU af[8];
    #pragma unroll
    for (int k8 = 0; k8 < 8; ++k8) af[k8] = load_frag_a(Fq, row0, k8 * 32, S_F, lane);

    float mv[8], rl[8];
    {
        const float* mp = mrow + row0 + half * 8;
        const float* lp = lrow + row0 + half * 8;
        #pragma unroll
        for (int r = 0; r < 8; ++r) { mv[r] = mp[r]; rl[r] = 1.0f / lp[r]; }
    }

    v8f acc[10];
    #pragma unroll
    for (int t = 0; t < 10; ++t)
        #pragma unroll
        for (int e = 0; e < 8; ++e) acc[t][e] = 0.f;

    u16* myP = sP[wave];

    for (int j = 0; j < L_TOK; j += 32) {
        __syncthreads();   // previous-iteration readers done
        // stage Fk rows j..j+31 (K=256)
        #pragma unroll
        for (int q = 0; q < 4; ++q) {
            int e = threadIdx.x + q * 256;
            int row = e >> 5, chunk = e & 31;
            *(u32x4*)&sFk[row * FK_LDS + chunk * 8] =
                *(const u32x4*)(Fk + (size_t)(j + row) * S_F + chunk * 8);
        }
        // stage Vt rows 0..319, cols j..j+31
        #pragma unroll
        for (int q = 0; q < 5; ++q) {
            int e = threadIdx.x + q * 256;        // 320*4 chunks
            int n = e >> 2, s = e & 3;
            *(u32x4*)&sVt[n * VT_LDS + s * 8] =
                *(const u32x4*)(Vt + (size_t)n * L_TOK + j + s * 8);
        }
        __syncthreads();

        // S tiles for columns j..j+15 and j+16..j+31
        v8f s0, s1;
        #pragma unroll
        for (int e = 0; e < 8; ++e) { s0[e] = 0.f; s1[e] = 0.f; }
        #pragma unroll
        for (int k8 = 0; k8 < 8; ++k8) {
            FragU b0 = lds_frag_b(sFk, 0,  k8 * 32, FK_LDS, lane);
            FragU b1 = lds_frag_b(sFk, 16, k8 * 32, FK_LDS, lane);
            s0 = wmma_bf16(af[k8], b0, s0);
            s1 = wmma_bf16(af[k8], b1, s1);
        }
        // P = exp(s - m) / l  -> per-wave LDS tile (row-major 16x32 bf16)
        #pragma unroll
        for (int r = 0; r < 8; ++r) {
            int rr = r + half * 8;
            myP[rr * 32 + l15]      = f2bf(__expf(s0[r] - mv[r]) * rl[r]);
            myP[rr * 32 + 16 + l15] = f2bf(__expf(s1[r] - mv[r]) * rl[r]);
        }
        __builtin_amdgcn_wave_barrier();   // same-wave LDS round trip: DS is in-order per wave
        FragU pa;
        {
            int k0 = half ? 16 : 0, k1 = half ? 24 : 8;
            pa.q[0] = *(const u32x4*)&myP[l15 * 32 + k0];
            pa.q[1] = *(const u32x4*)&myP[l15 * 32 + k1];
        }
        // acc += P(16x32) * V-block(32x160) from staged Vt
        #pragma unroll
        for (int t = 0; t < 10; ++t) {
            FragU b = lds_frag_b(sVt, nbase + t * 16, 0, VT_LDS, lane);
            acc[t] = wmma_bf16(pa, b, acc[t]);
        }
    }

    #pragma unroll
    for (int t = 0; t < 10; ++t) {
        int col = nbase + t * 16 + l15;
        #pragma unroll
        for (int r = 0; r < 8; ++r) {
            int row = row0 + r + half * 8;
            out[(size_t)row * S_EMB + col] = acc[t][r];
        }
    }
}

// ---------------- deterministic column sum ----------------
__global__ void colsum_partial(const float* __restrict__ v, float* __restrict__ part) {
    int t = threadIdx.x, b = blockIdx.x;         // 32 blocks x 256 rows each
    if (t >= 200) return;
    float s = 0.f;
    int rbase = b * 256;
    for (int r = 0; r < 256; ++r) s += v[(size_t)(rbase + r) * S_F + t];
    part[b * 256 + t] = s;
}

__global__ void colsum_final(const float* __restrict__ part, float* __restrict__ hxseg) {
    int t = threadIdx.x;
    if (t >= 200) return;
    float s = 0.f;
    #pragma unroll
    for (int b = 0; b < 32; ++b) s += part[b * 256 + t];
    hxseg[t] = s;
}

// ---------------- final H classifier + softmax ----------------
__global__ void final_kernel(const float* __restrict__ hx,
                             const float* __restrict__ Hw1, const float* __restrict__ Hb1,
                             const float* __restrict__ Hw2, const float* __restrict__ Hb2,
                             float* __restrict__ out) {
    __shared__ float sh[200];
    __shared__ float lg[3];
    int t = threadIdx.x;
    if (t < 200) {
        float s = Hb1[t];
        const float* w = Hw1 + (size_t)t * 400;
        for (int k = 0; k < 400; ++k) s += hx[k] * w[k];
        sh[t] = fmaxf(s, 0.f);
    }
    __syncthreads();
    if (t < 3) {
        float s = Hb2[t];
        const float* w = Hw2 + (size_t)t * 200;
        for (int k = 0; k < 200; ++k) s += sh[k] * w[k];
        lg[t] = s;
    }
    __syncthreads();
    if (t == 0) {
        float m = fmaxf(lg[0], fmaxf(lg[1], lg[2]));
        float e0 = __expf(lg[0] - m), e1 = __expf(lg[1] - m), e2 = __expf(lg[2] - m);
        float inv = 1.0f / (e0 + e1 + e2);
        out[0] = e0 * inv; out[1] = e1 * inv; out[2] = e2 * inv;
    }
}

// ---------------- host-side launcher ----------------
extern "C" void kernel_launch(void* const* d_in, const int* in_sizes, int n_in,
                              void* d_out, int out_size, void* d_ws, size_t ws_size,
                              hipStream_t stream) {
    const float* sen1 = (const float*)d_in[0];
    const float* sen2 = (const float*)d_in[1];
    const float* F_w1 = (const float*)d_in[2];
    const float* F_b1 = (const float*)d_in[3];
    const float* F_w2 = (const float*)d_in[4];
    const float* F_b2 = (const float*)d_in[5];
    const float* G_w1 = (const float*)d_in[6];
    const float* G_b1 = (const float*)d_in[7];
    const float* G_w2 = (const float*)d_in[8];
    const float* G_b2 = (const float*)d_in[9];
    const float* H_w1 = (const float*)d_in[10];
    const float* H_b1 = (const float*)d_in[11];
    const float* H_w2 = (const float*)d_in[12];
    const float* H_b2 = (const float*)d_in[13];

    char* ws = (char*)d_ws;
    size_t o = 0;
    auto alloc = [&](size_t bytes) { void* p = ws + o; o += (bytes + 255) & ~(size_t)255; return p; };

    u16*   sen1b = (u16*)alloc((size_t)L_TOK * S_EMB * 2);
    u16*   sen2b = (u16*)alloc((size_t)L_TOK * S_EMB * 2);
    u16*   s1T   = (u16*)alloc((size_t)S_EMB * L_TOK * 2);
    u16*   s2T   = (u16*)alloc((size_t)S_EMB * L_TOK * 2);
    u16*   Fw1b  = (u16*)alloc((size_t)NPAD * S_EMB * 2);
    u16*   Fw2b  = (u16*)alloc((size_t)NPAD * S_F   * 2);
    u16*   Gw1b  = (u16*)alloc((size_t)NPAD * S_CAT * 2);
    u16*   Gw2b  = (u16*)alloc((size_t)NPAD * S_F   * 2);
    u16*   Fa    = (u16*)alloc((size_t)L_TOK * S_F * 2);
    u16*   Fb    = (u16*)alloc((size_t)L_TOK * S_F * 2);
    u16*   htmp  = (u16*)alloc((size_t)L_TOK * S_F * 2);
    float* mr    = (float*)alloc(L_TOK * 4);
    float* lr    = (float*)alloc(L_TOK * 4);
    float* mc    = (float*)alloc(L_TOK * 4);
    float* lc    = (float*)alloc(L_TOK * 4);
    float* beta  = (float*)alloc((size_t)L_TOK * S_EMB * 4);
    float* alpha = (float*)alloc((size_t)L_TOK * S_EMB * 4);
    u16*   cat   = (u16*)alloc((size_t)L_TOK * S_CAT * 2);
    float* vbuf  = (float*)alloc((size_t)L_TOK * S_F * 4);
    float* part  = (float*)alloc(32 * 256 * 4);
    float* hx    = (float*)alloc(512 * 4);

    // 0) zero hx
    zero_f32<<<2, 256, 0, stream>>>(hx, 512);

    // 1) casts / transposes
    {
        int tot = L_TOK * S_EMB;
        cast_pad<<<(tot + 255) / 256, 256, 0, stream>>>(sen1, sen1b, L_TOK, EMB, S_EMB, tot);
        cast_pad<<<(tot + 255) / 256, 256, 0, stream>>>(sen2, sen2b, L_TOK, EMB, S_EMB, tot);
        transpose_cast<<<(tot + 255) / 256, 256, 0, stream>>>(sen1, s1T, L_TOK, EMB, tot);
        transpose_cast<<<(tot + 255) / 256, 256, 0, stream>>>(sen2, s2T, L_TOK, EMB, tot);
    }
    cast_pad<<<(NPAD * S_EMB + 255) / 256, 256, 0, stream>>>(F_w1, Fw1b, 200, 300, S_EMB, NPAD * S_EMB);
    cast_pad<<<(NPAD * S_F   + 255) / 256, 256, 0, stream>>>(F_w2, Fw2b, 200, 200, S_F,   NPAD * S_F);
    cast_pad<<<(NPAD * S_CAT + 255) / 256, 256, 0, stream>>>(G_w1, Gw1b, 200, 600, S_CAT, NPAD * S_CAT);
    cast_pad<<<(NPAD * S_F   + 255) / 256, 256, 0, stream>>>(G_w2, Gw2b, 200, 200, S_F,   NPAD * S_F);

    dim3 ggrid(L_TOK / 128, S_F / 32);   // 64 x 8
    // 2) F MLP: Fa = relu(relu(sen1*Fw1^T+b1)*Fw2^T+b2), same for Fb
    gemm_bias_relu<true><<<ggrid, 256, 0, stream>>>(sen1b, Fw1b, F_b1, htmp, S_EMB, S_EMB, S_EMB, S_F, 200);
    gemm_bias_relu<true><<<ggrid, 256, 0, stream>>>(htmp,  Fw2b, F_b2, Fa,   S_F,   S_F,   S_F,   S_F, 200);
    gemm_bias_relu<true><<<ggrid, 256, 0, stream>>>(sen2b, Fw1b, F_b1, htmp, S_EMB, S_EMB, S_EMB, S_F, 200);
    gemm_bias_relu<true><<<ggrid, 256, 0, stream>>>(htmp,  Fw2b, F_b2, Fb,   S_F,   S_F,   S_F,   S_F, 200);

    // 3) flash softmax stats: rows of E (Fa,Fb) and rows of E^T (Fb,Fa)
    softmax_stats<<<L_TOK / 128, 256, 0, stream>>>(Fa, Fb, mr, lr);
    softmax_stats<<<L_TOK / 128, 256, 0, stream>>>(Fb, Fa, mc, lc);

    // 4) beta = softmax_row(E) @ sen2 ; alpha = softmax_row(E^T) @ sen1
    attend_kernel<<<L_TOK / 64, 256, 0, stream>>>(Fa, Fb, mr, lr, s2T, beta);
    attend_kernel<<<L_TOK / 64, 256, 0, stream>>>(Fb, Fa, mc, lc, s1T, alpha);

    // 5) v1 = G(concat(sen1,beta)); column sums -> hx[0:200]
    concat_kernel<<<L_TOK, S_CAT, 0, stream>>>(sen1b, beta, cat);
    gemm_bias_relu<true ><<<ggrid, 256, 0, stream>>>(cat,  Gw1b, G_b1, htmp, S_CAT, S_CAT, S_CAT, S_F, 200);
    gemm_bias_relu<false><<<ggrid, 256, 0, stream>>>(htmp, Gw2b, G_b2, vbuf, S_F,   S_F,   S_F,   S_F, 200);
    colsum_partial<<<32, 256, 0, stream>>>(vbuf, part);
    colsum_final<<<1, 256, 0, stream>>>(part, hx);

    // 6) v2 = G(concat(sen2,alpha)); column sums -> hx[200:400]
    concat_kernel<<<L_TOK, S_CAT, 0, stream>>>(sen2b, alpha, cat);
    gemm_bias_relu<true ><<<ggrid, 256, 0, stream>>>(cat,  Gw1b, G_b1, htmp, S_CAT, S_CAT, S_CAT, S_F, 200);
    gemm_bias_relu<false><<<ggrid, 256, 0, stream>>>(htmp, Gw2b, G_b2, vbuf, S_F,   S_F,   S_F,   S_F, 200);
    colsum_partial<<<32, 256, 0, stream>>>(vbuf, part);
    colsum_final<<<1, 256, 0, stream>>>(part, hx + 200);

    // 7) H classifier + softmax -> d_out[3]
    final_kernel<<<1, 256, 0, stream>>>(hx, H_w1, H_b1, H_w2, H_b2, (float*)d_out);
}
